// TransformerWithModernAttention_24292335027127
// MI455X (gfx1250) — matrix operand order repaired and checked by reading, not compiled, not run on hardware
//
#include <hip/hip_runtime.h>
#include <hip/hip_bf16.h>
#include <math.h>

// ---------------------------------------------------------------------------
// Problem constants (from the reference)
// ---------------------------------------------------------------------------
namespace cfg {
constexpr int S   = 2048;         // sequence length
constexpr int Dm  = 1024;         // model dim
constexpr int Hh  = 16;           // heads
constexpr int HD  = 64;           // head dim
constexpr int DFF = 4096;         // ffn dim
constexpr int NS  = 8;            // sink tokens
constexpr int SA  = S + NS;       // sink-layer sequence length (2056)
}

// ---------------------------------------------------------------------------
// Types for CDNA5 WMMA (wave32): V_WMMA_F32_16X16X32_BF16
// ---------------------------------------------------------------------------
typedef __attribute__((ext_vector_type(16))) __bf16 v16bf;
typedef __attribute__((ext_vector_type(8)))  __bf16 v8bf;
typedef __attribute__((ext_vector_type(4)))  __bf16 v4bf;
typedef __attribute__((ext_vector_type(2)))  __bf16 v2bf;
typedef __attribute__((ext_vector_type(8)))  float  v8f;
typedef __attribute__((ext_vector_type(4)))  float  v4f;
typedef __attribute__((ext_vector_type(4)))  unsigned uint4v;
typedef __attribute__((ext_vector_type(8)))  int      int8v;
typedef __attribute__((ext_vector_type(4)))  int      int4v;

static __device__ __forceinline__ __bf16 f2bf(float f) { return (__bf16)f; }

static __device__ __forceinline__ v8bf ld8(const __bf16* p) {
    return *(const v8bf*)p;                       // 16B load (ds/global b128)
}
static __device__ __forceinline__ v16bf cat16(v8bf lo, v8bf hi) {
    return __builtin_shufflevector(lo, hi, 0, 1, 2, 3, 4, 5, 6, 7,
                                            8, 9, 10, 11, 12, 13, 14, 15);
}
// staging helpers: 8 elements global -> LDS (bf16 passthrough or fp32 convert)
static __device__ __forceinline__ void stage8(__bf16* dst, const __bf16* src) {
    *(v8bf*)dst = *(const v8bf*)src;
}
static __device__ __forceinline__ void stage8(__bf16* dst, const float* src) {
    v4f x = *(const v4f*)src;
    v4f y = *(const v4f*)(src + 4);
    v8bf t;
    t[0] = f2bf(x[0]); t[1] = f2bf(x[1]); t[2] = f2bf(x[2]); t[3] = f2bf(x[3]);
    t[4] = f2bf(y[0]); t[5] = f2bf(y[1]); t[6] = f2bf(y[2]); t[7] = f2bf(y[3]);
    *(v8bf*)dst = t;
}
static __device__ __forceinline__ void zero8(__bf16* dst) { *(v8bf*)dst = (v8bf){}; }

static __device__ __forceinline__ void store_out(float* p, float v)  { *p = v; }
static __device__ __forceinline__ void store_out(__bf16* p, float v) { *p = f2bf(v); }

static __device__ __forceinline__ v8f wmma_bf16(v16bf a, v16bf b, v8f c) {
    return __builtin_amdgcn_wmma_f32_16x16x32_bf16(false, a, false, b,
                                                   (short)0, c, false, false);
}

// ---------------------------------------------------------------------------
// Cross-lane helpers on v_permlane16_b32 (VALU, no LDS traffic)
// ---------------------------------------------------------------------------
static constexpr unsigned xsel_lo(int d) {
    unsigned s = 0;
    for (int i = 0; i < 8; ++i) s |= (unsigned)((i ^ d) & 15) << (4 * i);
    return s;
}
static constexpr unsigned xsel_hi(int d) {
    unsigned s = 0;
    for (int i = 8; i < 16; ++i) s |= (unsigned)((i ^ d) & 15) << (4 * (i - 8));
    return s;
}
template <int D>
static __device__ __forceinline__ float bfly16(float v) {
    unsigned u = __builtin_bit_cast(unsigned, v);
    u = __builtin_amdgcn_permlane16(u, u, xsel_lo(D), xsel_hi(D), false, false);
    return __builtin_bit_cast(float, u);
}
static __device__ __forceinline__ float bcast16(float v) {   // lane0 of each row
    unsigned u = __builtin_bit_cast(unsigned, v);
    u = __builtin_amdgcn_permlane16(u, u, 0u, 0u, false, false);
    return __builtin_bit_cast(float, u);
}
static __device__ __forceinline__ float rowmax16(float v) {
    v = fmaxf(v, bfly16<1>(v));
    v = fmaxf(v, bfly16<2>(v));
    v = fmaxf(v, bfly16<4>(v));
    v = fmaxf(v, bfly16<8>(v));
    return v;
}

// ---------------------------------------------------------------------------
// TDM: 2-D tile of bf16 rows (tile_k elems per row, tile_n rows) from global
// (row stride = stride_elems) into LDS at lds_byte_addr, with 16B padding
// after every 128B row => LDS row stride 72 bf16. Issued by one wave.
// Descriptor packing per CDNA5 ISA §8.3/8.4 (2-D: groups 2/3 zero).
// This toolchain exposes the 6-arg builtin (extra int32x8 before cpol).
// ---------------------------------------------------------------------------
static __device__ __forceinline__ void tdm_load_2d(unsigned lds_byte_addr,
                                                   const __bf16* gsrc,
                                                   unsigned tile_k, unsigned tile_n,
                                                   unsigned stride_elems) {
    unsigned long long ga = (unsigned long long)(size_t)gsrc;
    uint4v g0;
    g0[0] = 1u;                                         // count=1 user D#
    g0[1] = lds_byte_addr;                              // lds_addr
    g0[2] = (unsigned)(ga & 0xffffffffu);               // global_addr lo
    g0[3] = (unsigned)((ga >> 32) & 0x1ffffffu)         // global_addr hi (56:32)
          | (2u << 30);                                 // type = 2 ("image")
    int8v g1;
    g1[0] = (1 << 16)                                   // data_size = 2 bytes
          | (1 << 20)                                   // pad_enable
          | (4 << 22)                                   // pad_interval: 16x8B=128B
          | (3 << 25);                                  // pad_amount: 4 DW = 16B
    g1[1] = (int)(tile_k << 16);                        // tensor_dim0 lo (63:48)
    g1[2] = (int)((tile_k >> 16) | (tile_n << 16));     // dim0 hi | tensor_dim1 lo
    g1[3] = (int)((tile_n >> 16) | (tile_k << 16));     // dim1 hi | tile_dim0
    g1[4] = (int)tile_n;                                // tile_dim1 (tile_dim2=0)
    g1[5] = (int)stride_elems;                          // tensor_dim0_stride lo
    g1[6] = 0;                                          // stride0 hi | dim1_stride
    g1[7] = 0;
    int4v gz  = {0, 0, 0, 0};
    int8v gz8 = {0, 0, 0, 0, 0, 0, 0, 0};
    __builtin_amdgcn_tensor_load_to_lds(g0, g1, gz, gz, gz8, 0);
}

// ---------------------------------------------------------------------------
// GEMM:  C[M,N] = act( (A[M,K] * WT^T + bias) * scale )
// Weights pre-converted to bf16 and pre-transposed: WT[n][k].
// 64(M) x 128(N) tile, 8 waves, K-step 64 (8 WMMAs per wave per step).
// B tile fetched by the Tensor Data Mover (wave 0) while all waves stage A.
// TA in {float, __bf16}; TO in {float, __bf16}.
// ---------------------------------------------------------------------------
template <class TA, class TO, bool RELU>
__global__ __launch_bounds__(256)
void gemm_kernel(const TA* __restrict__ A, const __bf16* __restrict__ WT,
                 const float* __restrict__ bias, TO* __restrict__ C,
                 int M, int N, int K, float scale) {
    __shared__ __bf16 As [64][72];     // [m][k]  row stride 144B
    __shared__ __bf16 WsT[128][72];    // [n][k]  row stride 144B (TDM-padded)

    const int tid  = threadIdx.x;
    const int lane = tid & 31;
    const int hf   = lane >> 4;
    const int ln   = lane & 15;
    const int wid  = tid >> 5;
    const int wm   = wid >> 1;         // 0..3 (M subtile of 16)
    const int wn   = wid & 1;          // 0..1 (N subtile of 64)

    const int m0 = blockIdx.y * 64;
    const int n0 = blockIdx.x * 128;
    const unsigned wlds = (unsigned)(size_t)&WsT[0][0];   // LDS byte offset

    v8f acc[4] = {};

    for (int kt = 0; kt < K; kt += 64) {
        // B tile 128(n) x 64(k): async DMA into LDS (one wave issues)
        if (wid == 0) {
            tdm_load_2d(wlds, &WT[(size_t)n0 * K + kt], 64u, 128u, (unsigned)K);
        }
        // A tile 64x64: all waves stage (converted if fp32)
        for (int i = tid; i < 64 * 8; i += 256) {
            int r = i >> 3, c8 = (i & 7) * 8;
            int gm = m0 + r;
            if (gm < M) stage8(&As[r][c8], &A[(size_t)gm * K + kt + c8]);
            else        zero8(&As[r][c8]);
        }
        if (wid == 0) __builtin_amdgcn_s_wait_tensorcnt(0);
        __syncthreads();

        const int arow = wm * 16 + ln;
        v16bf a0 = cat16(ld8(&As[arow][hf * 16]),      ld8(&As[arow][hf * 16 + 8]));
        v16bf a1 = cat16(ld8(&As[arow][32 + hf * 16]), ld8(&As[arow][40 + hf * 16]));
#pragma unroll
        for (int t = 0; t < 4; ++t) {
            const __bf16* bp = &WsT[wn * 64 + t * 16 + ln][hf * 16];
            v16bf b0 = cat16(ld8(bp),      ld8(bp + 8));
            v16bf b1 = cat16(ld8(bp + 32), ld8(bp + 40));
            acc[t] = wmma_bf16(a0, b0, acc[t]);
            acc[t] = wmma_bf16(a1, b1, acc[t]);
        }
        __syncthreads();
    }

    // epilogue: C layout row = hf*8 + r, col = ln (per 16-wide tile)
#pragma unroll
    for (int t = 0; t < 4; ++t) {
#pragma unroll
        for (int r = 0; r < 8; ++r) {
            int gm = m0 + wm * 16 + hf * 8 + r;
            if (gm < M) {
                int gn = n0 + wn * 64 + t * 16 + ln;
                float v = (acc[t][r] + bias[gn]) * scale;
                if (RELU) v = fmaxf(v, 0.0f);
                store_out(&C[(size_t)gm * N + gn], v);
            }
        }
    }
}

// ---------------------------------------------------------------------------
// Weight pre-pass: WT[n][k] (bf16) = W[k][n] (fp32); one-time per layer.
// ---------------------------------------------------------------------------
__global__ __launch_bounds__(256)
void wt_bf16_kernel(const float* __restrict__ W, __bf16* __restrict__ WT,
                    int K, int N) {
    int idx = blockIdx.x * 256 + threadIdx.x;        // N * K/4 threads
    int n  = idx % N;
    int k4 = (idx / N) * 4;
    v4bf t;
    t[0] = f2bf(W[(size_t)(k4 + 0) * N + n]);
    t[1] = f2bf(W[(size_t)(k4 + 1) * N + n]);
    t[2] = f2bf(W[(size_t)(k4 + 2) * N + n]);
    t[3] = f2bf(W[(size_t)(k4 + 3) * N + n]);
    *(v4bf*)&WT[(size_t)n * K + k4] = t;
}

// ---------------------------------------------------------------------------
// Flash-style attention, all-bf16 streams (Q pre-scaled by 1/8 in its GEMM).
// One wave = 16 query rows of one head; block = 8 waves; 32-key tiles.
// Denominator accumulated by a 5th WMMA against a ones-column; row max via
// permlane16; P columns interleaved & packed, V LDS rows permuted to match.
// variant: 0 = ALiBi bias, 1 = sink (no bias), 2 = dilated mask.
// ---------------------------------------------------------------------------
__global__ __launch_bounds__(256)
void attn_kernel(const __bf16* __restrict__ Q, const __bf16* __restrict__ Kg,
                 const __bf16* __restrict__ Vg, __bf16* __restrict__ Out,
                 int Sq, int Skv, int qoff, int variant) {
    using namespace cfg;
    (void)Sq;
    __shared__ __bf16 Ks [32][72];     // [key][hd]         row stride 144B
    __shared__ __bf16 VsT[64][40];     // [hd][perm(key)]   row stride 80B
    __shared__ __bf16 Ps[8][16][40];   // per-wave P (interleaved columns)

    const int tid  = threadIdx.x;
    const int lane = tid & 31;
    const int hf   = lane >> 4;
    const int ln   = lane & 15;
    const int wid  = tid >> 5;
    const int h    = blockIdx.y;
    const int qbase = blockIdx.x * 128 + wid * 16;

    // wave's Q tile (16x64) -> two A fragments (16B global loads, no convert)
    v16bf a0, a1;
    {
        const __bf16* qp = Q + (size_t)(qoff + qbase + ln) * Dm + h * HD;
        a0 = cat16(ld8(qp + hf * 16),      ld8(qp + 8 + hf * 16));
        a1 = cat16(ld8(qp + 32 + hf * 16), ld8(qp + 40 + hf * 16));
    }

    // ones-column B operand: B[k][n] = (n==0) ? 1 : 0  (permutation-invariant)
    v16bf bone;
    {
        __bf16 one = (ln == 0) ? (__bf16)1.0f : (__bf16)0.0f;
#pragma unroll
        for (int e = 0; e < 16; ++e) bone[e] = one;
    }

    float rm[8];
#pragma unroll
    for (int r = 0; r < 8; ++r) rm[r] = -INFINITY;
    v8f o0 = {}, o1 = {}, o2 = {}, o3 = {}, o4 = {};   // o4 = denominator

    const float slope = -exp2f((float)h - 8.0f);   // ALiBi slope
    const int   rmask = (1 << (h & 3)) - 1;        // dilation rate - 1

    for (int kt = 0; kt < Skv; kt += 32) {
        // stage K (row-major, 16B copies) and V (transposed, key interleaved)
        for (int i = tid; i < 32 * 8; i += 256) {
            int r = i >> 3, c8 = (i & 7) * 8;
            int kp = kt + r;
            if (kp < Skv) stage8(&Ks[r][c8], &Kg[(size_t)kp * Dm + h * HD + c8]);
            else          zero8(&Ks[r][c8]);
            v8bf vv = {};
            if (kp < Skv) vv = ld8(&Vg[(size_t)kp * Dm + h * HD + c8]);
            int rp = ((r & 15) << 1) | (r >> 4);       // perm(key) = interleave
#pragma unroll
            for (int j = 0; j < 8; ++j) VsT[c8 + j][rp] = vv[j];
        }
        if (kt + 32 < Skv) {   // global_prefetch of the next K/V tiles
            __builtin_prefetch(Kg + (size_t)(kt + 32) * Dm + h * HD + tid, 0, 1);
            __builtin_prefetch(Vg + (size_t)(kt + 32) * Dm + h * HD + tid, 0, 1);
        }
        __syncthreads();

        // scores: two 16x16 C tiles = Q(16x64) * K^T(64x32)
        v8f s0 = {}, s1 = {};
        {
            const __bf16* k0 = &Ks[ln][hf * 16];        // keys 0..15
            const __bf16* k1 = &Ks[16 + ln][hf * 16];   // keys 16..31
            v16bf b00 = cat16(ld8(k0), ld8(k0 + 8));
            v16bf b10 = cat16(ld8(k0 + 32), ld8(k0 + 40));
            v16bf b01 = cat16(ld8(k1), ld8(k1 + 8));
            v16bf b11 = cat16(ld8(k1 + 32), ld8(k1 + 40));
            s0 = wmma_bf16(a0, b00, s0); s0 = wmma_bf16(a1, b10, s0);
            s1 = wmma_bf16(a0, b01, s1); s1 = wmma_bf16(a1, b11, s1);
        }

        const int kc0 = kt + ln, kc1 = kt + 16 + ln;

        // bias/mask + online softmax (rows live across 16-lane half groups)
#pragma unroll
        for (int r = 0; r < 8; ++r) {
            int qp = qbase + hf * 8 + r;
            float sv0 = s0[r], sv1 = s1[r];
            if (variant == 0) {            // ALiBi: rel * slope
                sv0 += -fabsf((float)(kc0 - qp)) * slope;
                sv1 += -fabsf((float)(kc1 - qp)) * slope;
            } else if (variant == 2) {     // dilated: keep iff |i-j| % rate == 0
                int d0 = kc0 - qp; d0 = d0 < 0 ? -d0 : d0;
                int d1 = kc1 - qp; d1 = d1 < 0 ? -d1 : d1;
                if (d0 & rmask) sv0 = -1e9f;
                if (d1 & rmask) sv1 = -1e9f;
            }
            if (kc0 >= Skv) sv0 = -1e9f;
            if (kc1 >= Skv) sv1 = -1e9f;

            float mnew  = fmaxf(rm[r], rowmax16(fmaxf(sv0, sv1)));
            float alpha = __expf(rm[r] - mnew);
            float p0 = __expf(sv0 - mnew);
            float p1 = __expf(sv1 - mnew);
            rm[r] = mnew;
            o0[r] *= alpha; o1[r] *= alpha; o2[r] *= alpha; o3[r] *= alpha;
            o4[r] *= alpha;

            v2bf pk;                       // packed pair -> interleaved columns
            pk[0] = f2bf(p0); pk[1] = f2bf(p1);
            *(v2bf*)&Ps[wid][hf * 8 + r][2 * ln] = pk;
        }

        // P(16x32) * V(32x64) + denominator tile (P * ones-column)
        {
            const __bf16* pp = &Ps[wid][ln][0];
            v16bf ap = cat16(ld8(pp + hf * 16), ld8(pp + 8 + hf * 16));
#pragma unroll
            for (int t = 0; t < 4; ++t) {
                const __bf16* vp = &VsT[t * 16 + ln][hf * 16];
                v16bf bv = cat16(ld8(vp), ld8(vp + 8));
                if      (t == 0) o0 = wmma_bf16(ap, bv, o0);
                else if (t == 1) o1 = wmma_bf16(ap, bv, o1);
                else if (t == 2) o2 = wmma_bf16(ap, bv, o2);
                else             o3 = wmma_bf16(ap, bv, o3);
            }
            o4 = wmma_bf16(ap, bone, o4);  // row sums accumulate in column 0
        }
        __syncthreads();
    }

    // normalize by the accumulated denominator; bf16 out feeds the O-proj GEMM
#pragma unroll
    for (int r = 0; r < 8; ++r) {
        float inv = 1.0f / bcast16(o4[r]);
        int gq = qbase + hf * 8 + r;
        __bf16* op = Out + (size_t)gq * Dm + h * HD;
        op[ln]      = f2bf(o0[r] * inv);
        op[16 + ln] = f2bf(o1[r] * inv);
        op[32 + ln] = f2bf(o2[r] * inv);
        op[48 + ln] = f2bf(o3[r] * inv);
    }
}

// ---------------------------------------------------------------------------
// out[row] = LayerNorm(x[row] + f[row]) * g + b   (one block per row, fp32)
// ---------------------------------------------------------------------------
__global__ __launch_bounds__(256)
void add_ln_kernel(const float* __restrict__ x, const float* __restrict__ f,
                   const float* __restrict__ g, const float* __restrict__ b,
                   float* __restrict__ out) {
    using namespace cfg;
    __shared__ float red[16];
    const int row = blockIdx.x;
    const int tid = threadIdx.x;
    const int c0  = tid * 4;

    v4f xv = *(const v4f*)&x[(size_t)row * Dm + c0];
    v4f fv = *(const v4f*)&f[(size_t)row * Dm + c0];
    float v[4];
    float s = 0.0f, ss = 0.0f;
#pragma unroll
    for (int j = 0; j < 4; ++j) {
        float t = xv[j] + fv[j];
        v[j] = t; s += t; ss += t * t;
    }
#pragma unroll
    for (int d = 1; d < 32; d <<= 1) { s += __shfl_xor(s, d, 32); ss += __shfl_xor(ss, d, 32); }
    if ((tid & 31) == 0) { red[tid >> 5] = s; red[8 + (tid >> 5)] = ss; }
    __syncthreads();
    float ts = 0.0f, tss = 0.0f;
#pragma unroll
    for (int i = 0; i < 8; ++i) { ts += red[i]; tss += red[8 + i]; }
    float mean = ts * (1.0f / Dm);
    float var  = tss * (1.0f / Dm) - mean * mean;
    float rstd = rsqrtf(var + 1e-5f);
    v4f gv = *(const v4f*)&g[c0];
    v4f bv = *(const v4f*)&b[c0];
    v4f ov;
#pragma unroll
    for (int j = 0; j < 4; ++j) ov[j] = (v[j] - mean) * rstd * gv[j] + bv[j];
    *(v4f*)&out[(size_t)row * Dm + c0] = ov;
}

// xa = concat(sink[NS,D], x[S,D]) — vectorized copy
__global__ __launch_bounds__(256)
void build_xa_kernel(const float* __restrict__ sink, const float* __restrict__ x,
                     float* __restrict__ xa) {
    using namespace cfg;
    int i4 = (blockIdx.x * 256 + threadIdx.x) * 4;
    int row = i4 / Dm;
    v4f v = (row < NS) ? *(const v4f*)&sink[i4] : *(const v4f*)&x[i4 - NS * Dm];
    *(v4f*)&xa[i4] = v;
}

// ---------------------------------------------------------------------------
// Host side
// ---------------------------------------------------------------------------
struct Layer {
    const float *wq, *bq, *wk, *bk, *wv, *bv, *wo, *bo;
    const float *wf1, *bf1, *wf2, *bf2, *g1, *be1, *g2, *be2;
};

extern "C" void kernel_launch(void* const* d_in, const int* in_sizes, int n_in,
                              void* d_out, int out_size, void* d_ws, size_t ws_size,
                              hipStream_t stream) {
    using namespace cfg;
    (void)in_sizes; (void)n_in; (void)out_size; (void)ws_size;

    int idx = 0;
    const float* x = (const float*)d_in[idx++];
    auto grab = [&](Layer& L) {
        L.wq  = (const float*)d_in[idx++]; L.bq  = (const float*)d_in[idx++];
        L.wk  = (const float*)d_in[idx++]; L.bk  = (const float*)d_in[idx++];
        L.wv  = (const float*)d_in[idx++]; L.bv  = (const float*)d_in[idx++];
        L.wo  = (const float*)d_in[idx++]; L.bo  = (const float*)d_in[idx++];
        L.wf1 = (const float*)d_in[idx++]; L.bf1 = (const float*)d_in[idx++];
        L.wf2 = (const float*)d_in[idx++]; L.bf2 = (const float*)d_in[idx++];
        L.g1  = (const float*)d_in[idx++]; L.be1 = (const float*)d_in[idx++];
        L.g2  = (const float*)d_in[idx++]; L.be2 = (const float*)d_in[idx++];
    };
    Layer L1; grab(L1);
    Layer L2; grab(L2);
    const float* sink = (const float*)d_in[idx++];   // p2['sink'] appended last
    Layer L3; grab(L3);

    // workspace layout (fp32 regions first, then bf16; all 16B aligned)
    char* p = (char*)d_ws;
    auto alloc_f = [&](size_t n) { float*  r = (float*)p;  p += n * 4; return r; };
    auto alloc_b = [&](size_t n) { __bf16* r = (__bf16*)p; p += n * 2; return r; };
    float*  xa    = alloc_f((size_t)SA * Dm);
    float*  proj  = alloc_f((size_t)S * Dm);
    float*  bufA  = alloc_f((size_t)S * Dm);
    float*  bufB  = alloc_f((size_t)S * Dm);
    __bf16* qb    = alloc_b((size_t)SA * Dm);
    __bf16* kb    = alloc_b((size_t)SA * Dm);
    __bf16* vb    = alloc_b((size_t)SA * Dm);
    __bf16* attnb = alloc_b((size_t)S * Dm);
    __bf16* ffhb  = alloc_b((size_t)S * DFF);
    __bf16* wqT   = alloc_b((size_t)Dm * Dm);
    __bf16* wkT   = alloc_b((size_t)Dm * Dm);
    __bf16* wvT   = alloc_b((size_t)Dm * Dm);
    __bf16* woT   = alloc_b((size_t)Dm * Dm);
    __bf16* f1T   = alloc_b((size_t)Dm * DFF);
    __bf16* f2T   = alloc_b((size_t)DFF * Dm);

    auto wconv = [&](const float* W, __bf16* WT, int K, int N) {
        wt_bf16_kernel<<<((size_t)N * K / 4) / 256, 256, 0, stream>>>(W, WT, K, N);
    };
    auto gemm_bf = [&](const float* A, const __bf16* WT, const float* bias,
                       __bf16* C, int M, int N, int K, float scale, bool relu) {
        dim3 g(N / 128, (M + 63) / 64);
        if (relu) gemm_kernel<float, __bf16, true ><<<g, 256, 0, stream>>>(A, WT, bias, C, M, N, K, scale);
        else      gemm_kernel<float, __bf16, false><<<g, 256, 0, stream>>>(A, WT, bias, C, M, N, K, scale);
    };
    auto gemm_f = [&](const __bf16* A, const __bf16* WT, const float* bias,
                      float* C, int M, int N, int K) {
        dim3 g(N / 128, (M + 63) / 64);
        gemm_kernel<__bf16, float, false><<<g, 256, 0, stream>>>(A, WT, bias, C, M, N, K, 1.0f);
    };

    auto run_block = [&](const float* xin, float* xout, const Layer& L, int variant) {
        // one-time per layer: bf16 + transposed weights
        wconv(L.wq, wqT, Dm, Dm);   wconv(L.wk, wkT, Dm, Dm);
        wconv(L.wv, wvT, Dm, Dm);   wconv(L.wo, woT, Dm, Dm);
        wconv(L.wf1, f1T, Dm, DFF); wconv(L.wf2, f2T, DFF, Dm);

        const float* gin = xin;
        int Mq = S;
        if (variant == 1) {
            build_xa_kernel<<<(SA * Dm / 4) / 256, 256, 0, stream>>>(sink, xin, xa);
            gin = xa; Mq = SA;
        }
        gemm_bf(gin, wqT, L.bq, qb, Mq, Dm, Dm, 0.125f, false);  // Q pre-scaled
        gemm_bf(gin, wkT, L.bk, kb, Mq, Dm, Dm, 1.0f,  false);
        gemm_bf(gin, wvT, L.bv, vb, Mq, Dm, Dm, 1.0f,  false);

        dim3 ga(S / 128, Hh);
        attn_kernel<<<ga, 256, 0, stream>>>(qb, kb, vb, attnb, S, Mq,
                                            (variant == 1) ? NS : 0, variant);

        gemm_f(attnb, woT, L.bo, proj, S, Dm, Dm);
        add_ln_kernel<<<S, 256, 0, stream>>>(xin, proj, L.g1, L.be1, bufA);
        gemm_bf(bufA, f1T, L.bf1, ffhb, S, DFF, Dm, 1.0f, true);
        gemm_f(ffhb, f2T, L.bf2, proj, S, Dm, DFF);
        add_ln_kernel<<<S, 256, 0, stream>>>(bufA, proj, L.g2, L.be2, xout);
    };

    run_block(x,    bufB,          L1, 0);   // ALiBi
    run_block(bufB, bufB,          L2, 1);   // sink tokens
    run_block(bufB, (float*)d_out, L3, 2);   // dilated
}
